// TemporalLinearAttn_13683765805778
// MI455X (gfx1250) — matrix operand order repaired and checked by reading, not compiled, not run on hardware
//
#include <hip/hip_runtime.h>

// ---- problem constants (reference: B=2, T=16, C=256, H=64, W=64, heads=4) ----
#define T_DIM 16
#define C_DIM 256
#define N_HEADS 4
#define PPB 4            // pixels per block (consecutive along W)
#define NTHREADS 128     // 4 waves, one wave per pixel

typedef __attribute__((ext_vector_type(16))) __bf16 v16bf;
typedef __attribute__((ext_vector_type(8)))  __bf16 v8bf;
typedef __attribute__((ext_vector_type(8)))  float  v8f;

union Frag {
    v16bf        v;
    v8bf         h[2];
    unsigned int u[8];
};

__device__ __forceinline__ v8f wmma_bf16(const Frag& a, const Frag& b, v8f c) {
    return __builtin_amdgcn_wmma_f32_16x16x32_bf16(false, a.v, false, b.v,
                                                   (short)0, c, false, false);
}

__device__ __forceinline__ v8f zero8f() {
    v8f z;
#pragma unroll
    for (int i = 0; i < 8; ++i) z[i] = 0.0f;
    return z;
}

// Async DMA: global memory -> LDS, 16 bytes, tracked by ASYNCcnt (CDNA5).
__device__ __forceinline__ void async_load_b128(unsigned int lds_off,
                                                const void* gptr) {
    unsigned long long ga = (unsigned long long)gptr;
    asm volatile("global_load_async_to_lds_b128 %0, %1, off"
                 :: "v"(lds_off), "v"(ga) : "memory");
}

__device__ __forceinline__ void wait_asynccnt0() {
    asm volatile("s_wait_asynccnt 0x0" ::: "memory");
}

// B-fragment fetch from pre-swizzled bf16 weights: 32 contiguous bytes per lane.
__device__ __forceinline__ Frag loadBW(const __bf16* __restrict__ wb,
                                       int proj, int kk, int j, int lane) {
    Frag f;
    const __bf16* src = wb + (size_t)proj * 65536 + (((kk * 16 + j) * 32 + lane) << 4);
    f.h[0] = *(const v8bf*)(src);
    f.h[1] = *(const v8bf*)(src + 8);
    return f;
}

// ---- LDS layout (dynamic, 128 KB total) ----
// QS  [p][16][256] bf16  : q_soft, later attention-out          @ 0      (32 KB)
// XL  [p][16][256] bf16  : X in bf16, later kv[d][c] per head   @ 32768  (32 KB)
// XF  [16][256][p] f32   : async-staged raw X (dead after cvt)  @ 65536  (64 KB)
// KT  [p][4][64][16] bf16: k_soft transposed [head][c][t]       @ 65536  (32 KB)
// VT  [p][256][16] bf16  : v transposed [d][t]                  @ 98304  (32 KB)
// FO  [16][256][p] f32   : final staging (overlaps XL+KT, dead) @ 32768  (64 KB)
#define QS_OFF 0
#define XL_OFF 32768
#define XF_OFF 65536
#define KT_OFF 65536
#define VT_OFF 98304
#define FO_OFF 32768
#define SMEM_BYTES 131072

__global__ void __launch_bounds__(NTHREADS)
convert_weights(const float* __restrict__ Wq, const float* __restrict__ Wk,
                const float* __restrict__ Wv, const float* __restrict__ Wp,
                __bf16* __restrict__ wb) {
    const int proj = blockIdx.y;
    const float* W = (proj == 0) ? Wq : (proj == 1) ? Wk : (proj == 2) ? Wv : Wp;
    for (int idx = blockIdx.x * NTHREADS + threadIdx.x; idx < 65536;
         idx += gridDim.x * NTHREADS) {
        const int d = idx >> 8;
        const int c = idx & 255;
        const int j  = d >> 4;
        const int kk = c >> 5;
        const int lane = (d & 15) | (((c >> 3) & 1) << 4);  // hi lanes take K=8..15,24..31
        const int e = (c & 7) | (((c >> 4) & 1) << 3);
        wb[(size_t)proj * 65536 + (((kk * 16 + j) * 32 + lane) << 4) + e] = (__bf16)W[idx];
    }
}

__global__ void __launch_bounds__(NTHREADS)
attn_kernel(const float* __restrict__ x, const __bf16* __restrict__ wb,
            float* __restrict__ out, int Bsz, int H, int W) {
    extern __shared__ __align__(16) char smem[];

    const int tid  = threadIdx.x;
    const int lane = tid & 31;
    const int wave = tid >> 5;
    const int hi   = lane >> 4;     // 0: lanes 0-15, 1: lanes 16-31
    const int mrow = lane & 15;

    const int n0  = blockIdx.x * PPB;       // PPB consecutive pixels along W
    const int HW  = H * W;
    const int b   = n0 / HW;
    const int rem = n0 - b * HW;
    const int hh  = rem / W;
    const int w0  = rem - hh * W;           // multiple of PPB (PPB | W)

    // Warm the weight stream (global_prefetch) while staging X.
    __builtin_prefetch(wb + (tid << 4), 0, 0);

    // -------- Phase A1: async DMA X tile into LDS (fp32, [t][c][p]) ----------
    for (int idx = tid; idx < T_DIM * C_DIM; idx += NTHREADS) {
        const int t = idx >> 8;
        const int c = idx & 255;
        const float* gp = x + ((((size_t)b * T_DIM + t) * C_DIM + c) * H + hh) * W + w0;
        async_load_b128((unsigned int)(XF_OFF + (idx << 4)), gp);
    }
    wait_asynccnt0();
    __syncthreads();

    // -------- Phase A2: cooperative fp32 -> bf16 convert into XL [p][t][c] ----
    {
        const float* XF = (const float*)(smem + XF_OFF);
        for (int idx = tid; idx < T_DIM * C_DIM; idx += NTHREADS) {
            const float4 vv = *(const float4*)(XF + (idx << 2));
            __bf16* dst = (__bf16*)(smem + XL_OFF) + idx;   // idx == t*256+c
            dst[0]        = (__bf16)vv.x;
            dst[1 * 4096] = (__bf16)vv.y;
            dst[2 * 4096] = (__bf16)vv.z;
            dst[3 * 4096] = (__bf16)vv.w;
        }
    }
    __syncthreads();   // XF dead from here; KT/VT regions become usable

    const int p = wave;
    __bf16* Xp  = (__bf16*)(smem + XL_OFF) + p * 4096;  // [16][256]
    __bf16* Qp  = (__bf16*)(smem + QS_OFF) + p * 4096;  // [16][256]
    __bf16* KTp = (__bf16*)(smem + KT_OFF) + p * 4096;  // [4][64][16]
    __bf16* VTp = (__bf16*)(smem + VT_OFF) + p * 4096;  // [256][16]
    __bf16* KVp = Xp;                                   // reused: kv [64 d][64 c]

    // ---------------- A fragments of X (shared by Q/K/V projections) -----------
    Frag Ax[8];
#pragma unroll
    for (int kk = 0; kk < 8; ++kk) {
        const int c0 = kk * 32 + hi * 8;   // lane holds K {c0..c0+7} u {c0+16..c0+23}
        Ax[kk].h[0] = *(const v8bf*)(Xp + mrow * 256 + c0);
        Ax[kk].h[1] = *(const v8bf*)(Xp + mrow * 256 + c0 + 16);
    }

    const float scale = 0.125f;  // Ch^-0.5, Ch=64

    // ---------------- Q projection + softmax over head channels ----------------
    for (int head = 0; head < N_HEADS; ++head) {
        v8f acc[4];
#pragma unroll
        for (int jj = 0; jj < 4; ++jj) acc[jj] = zero8f();
#pragma unroll
        for (int jj = 0; jj < 4; ++jj) {
            const int j = head * 4 + jj;
#pragma unroll
            for (int kk = 0; kk < 8; ++kk)
                acc[jj] = wmma_bf16(Ax[kk], loadBW(wb, 0, kk, j, lane), acc[jj]);
        }
#pragma unroll
        for (int r = 0; r < 8; ++r) {
            float x0 = acc[0][r] * scale, x1 = acc[1][r] * scale;
            float x2 = acc[2][r] * scale, x3 = acc[3][r] * scale;
            float mx = fmaxf(fmaxf(x0, x1), fmaxf(x2, x3));
#pragma unroll
            for (int off = 1; off < 16; off <<= 1)
                mx = fmaxf(mx, __shfl_xor(mx, off, 32));
            float e0 = __expf(x0 - mx), e1 = __expf(x1 - mx);
            float e2 = __expf(x2 - mx), e3 = __expf(x3 - mx);
            float s = e0 + e1 + e2 + e3;
#pragma unroll
            for (int off = 1; off < 16; off <<= 1)
                s += __shfl_xor(s, off, 32);
            const float inv = 1.0f / s;
            __bf16* q = Qp + (r + 8 * hi) * 256 + head * 64 + mrow;
            q[0]  = (__bf16)(e0 * inv);
            q[16] = (__bf16)(e1 * inv);
            q[32] = (__bf16)(e2 * inv);
            q[48] = (__bf16)(e3 * inv);
        }
    }

    // ---------------- K projection + softmax over time, store transposed -------
    for (int j = 0; j < 16; ++j) {
        v8f acc = zero8f();
#pragma unroll
        for (int kk = 0; kk < 8; ++kk)
            acc = wmma_bf16(Ax[kk], loadBW(wb, 1, kk, j, lane), acc);
        float mx = acc[0];
#pragma unroll
        for (int r = 1; r < 8; ++r) mx = fmaxf(mx, acc[r]);
        mx = fmaxf(mx, __shfl_xor(mx, 16, 32));
        float e[8], s = 0.0f;
#pragma unroll
        for (int r = 0; r < 8; ++r) { e[r] = __expf(acc[r] - mx); s += e[r]; }
        s += __shfl_xor(s, 16, 32);
        const float inv = 1.0f / s;
        __bf16* kd = KTp + (j >> 2) * 1024 + ((j & 3) * 16 + mrow) * 16 + hi * 8;
#pragma unroll
        for (int r = 0; r < 8; ++r) kd[r] = (__bf16)(e[r] * inv);  // t = r+8*hi
    }

    // ---------------- V projection, store transposed [d][t] --------------------
    for (int j = 0; j < 16; ++j) {
        v8f acc = zero8f();
#pragma unroll
        for (int kk = 0; kk < 8; ++kk)
            acc = wmma_bf16(Ax[kk], loadBW(wb, 2, kk, j, lane), acc);
        __bf16* vd = VTp + (j * 16 + mrow) * 16 + hi * 8;
#pragma unroll
        for (int r = 0; r < 8; ++r) vd[r] = (__bf16)acc[r];
    }

    // ---------------- per-head linear attention: kv then out -------------------
    for (int head = 0; head < N_HEADS; ++head) {
        // A = k_softT [c][t], K = t (only 0..15 valid -> upper K half zero)
        Frag Ak[4];
#pragma unroll
        for (int i = 0; i < 4; ++i) {
            Ak[i].h[0] = *(const v8bf*)(KTp + head * 1024 + (i * 16 + mrow) * 16 + hi * 8);
#pragma unroll
            for (int u = 4; u < 8; ++u) Ak[i].u[u] = 0u;
        }
        // kv[c][d] = sum_t ksT[c][t] * vT[d][t]; store as kvT[d][c] in KVp
        for (int j = 0; j < 4; ++j) {
            Frag Bv;
            {  // lanes hold K=8*hi..8*hi+7 (=t) in elems 0-7; elems 8-15 (K=16..31) -> 0
                const __bf16* vr = VTp + (head * 64 + j * 16 + mrow) * 16 + hi * 8;
                Bv.h[0] = *(const v8bf*)(vr);
#pragma unroll
                for (int u = 4; u < 8; ++u) Bv.u[u] = 0u;
            }
#pragma unroll
            for (int i = 0; i < 4; ++i) {
                v8f kv = wmma_bf16(Ak[i], Bv, zero8f());
                __bf16* kvd = KVp + (j * 16 + mrow) * 64 + i * 16 + hi * 8;
#pragma unroll
                for (int r = 0; r < 8; ++r) kvd[r] = (__bf16)kv[r];
            }
        }
        // out[t][d] = sum_c q_soft[t][c] * kv[c][d]
        Frag Aq[2];
#pragma unroll
        for (int kk2 = 0; kk2 < 2; ++kk2) {
            const int c0 = head * 64 + kk2 * 32 + hi * 8;
            Aq[kk2].h[0] = *(const v8bf*)(Qp + mrow * 256 + c0);
            Aq[kk2].h[1] = *(const v8bf*)(Qp + mrow * 256 + c0 + 16);
        }
        v8f oacc[4];
#pragma unroll
        for (int j = 0; j < 4; ++j) oacc[j] = zero8f();
#pragma unroll
        for (int j = 0; j < 4; ++j) {
#pragma unroll
            for (int kk2 = 0; kk2 < 2; ++kk2) {
                Frag Bkv;
                const __bf16* kvr = KVp + (j * 16 + mrow) * 64 + kk2 * 32 + hi * 8;
                Bkv.h[0] = *(const v8bf*)(kvr);
                Bkv.h[1] = *(const v8bf*)(kvr + 16);
                oacc[j] = wmma_bf16(Aq[kk2], Bkv, oacc[j]);
            }
        }
        // overwrite this head's q_soft columns with the attention output
#pragma unroll
        for (int j = 0; j < 4; ++j)
#pragma unroll
            for (int r = 0; r < 8; ++r)
                Qp[(r + 8 * hi) * 256 + head * 64 + j * 16 + mrow] = (__bf16)oacc[j][r];
    }

    __syncthreads();  // XL/KT/VT now dead everywhere; FO may be written

    // ---------------- final projection with Wp, stage fp32 result --------------
    Frag Ao[8];
#pragma unroll
    for (int kk = 0; kk < 8; ++kk) {
        const int c0 = kk * 32 + hi * 8;
        Ao[kk].h[0] = *(const v8bf*)(Qp + mrow * 256 + c0);
        Ao[kk].h[1] = *(const v8bf*)(Qp + mrow * 256 + c0 + 16);
    }
    float* Fout = (float*)(smem + FO_OFF);  // [t][d][p]
    for (int j = 0; j < 16; ++j) {
        v8f acc = zero8f();
#pragma unroll
        for (int kk = 0; kk < 8; ++kk)
            acc = wmma_bf16(Ao[kk], loadBW(wb, 3, kk, j, lane), acc);
#pragma unroll
        for (int r = 0; r < 8; ++r)
            Fout[(((r + 8 * hi) * 256 + j * 16 + mrow) << 2) + p] = acc[r];
    }
    __syncthreads();

    // ---------------- coalesced float4 writeout over the 4 pixels --------------
    for (int idx = tid; idx < T_DIM * C_DIM; idx += NTHREADS) {
        const int t = idx >> 8;
        const int d = idx & 255;
        const float4 vv = *(const float4*)(Fout + ((t * 256 + d) << 2));
        *(float4*)(out + ((((size_t)b * T_DIM + t) * C_DIM + d) * H + hh) * W + w0) = vv;
    }
}

extern "C" void kernel_launch(void* const* d_in, const int* in_sizes, int n_in,
                              void* d_out, int out_size, void* d_ws, size_t ws_size,
                              hipStream_t stream) {
    const float* x  = (const float*)d_in[0];
    const float* Wq = (const float*)d_in[1];
    const float* Wk = (const float*)d_in[2];
    const float* Wv = (const float*)d_in[3];
    const float* Wp = (const float*)d_in[4];
    float* out = (float*)d_out;
    __bf16* wb = (__bf16*)d_ws;   // 4 * 256*256 * 2B = 512 KB, swizzled bf16 weights

    convert_weights<<<dim3(128, 4), NTHREADS, 0, stream>>>(Wq, Wk, Wv, Wp, wb);

    const int Bsz = 2, H = 64, W = 64;
    const int nblocks = (Bsz * H * W) / PPB;  // 2048
    attn_kernel<<<nblocks, NTHREADS, SMEM_BYTES, stream>>>(x, wb, out, Bsz, H, W);
}
